// XFeatDetector_33371895890585
// MI455X (gfx1250) — matrix operand order repaired and checked by reading, not compiled, not run on hardware
//
#include <hip/hip_runtime.h>

typedef _Float16 f16;
typedef _Float16 v16h __attribute__((ext_vector_type(16)));
typedef _Float16 v8h  __attribute__((ext_vector_type(8)));
typedef float    v8f  __attribute__((ext_vector_type(8)));

#define BNSCALE 0.9999950000374997f   /* 1/sqrt(1 + 1e-5) */
#define NKPT 4096
#define TOPK_N 16384

struct ConvDesc {
  int Hin, Win, Cin, Hout, Wout, Cout;
  int KH, KW, stride, pad;
  int Kreal, Kpad;
  int hasG, hasCB, hasBT, relu;
};

// ---------------------------------------------------------------------------
// Implicit-GEMM conv via V_WMMA_F32_16X16X32_F16, register-blocked over NB
// output-channel tiles: one wave owns 16 pixels x (NB*16) channels, loads the
// A (activation) fragment ONCE per K-chunk and issues NB independent WMMAs
// against it (pipelineable XDL ops, NB-x arithmetic intensity on activations).
// Activations NHWC f16; weights prepacked [Cout][Kpad] K-major f16
// (K = (ky*KW+kx)*Cin + ci), zero-padded to a multiple of 32. When
// Cin % 32 == 0 an aligned K-chunk sits inside one spatial tap, so both A and
// B fragments are contiguous 16B global loads (whole net is L2-resident).
// ---------------------------------------------------------------------------
template <int NB>
__global__ __launch_bounds__(32) void conv_wmma(
    const f16* __restrict__ act, const f16* __restrict__ wpk,
    const float* __restrict__ gamma, const float* __restrict__ beta,
    const float* __restrict__ cbias,
    f16* __restrict__ out16, float* __restrict__ out32, ConvDesc d)
{
  const int lane  = threadIdx.x;
  const long long mbase = (long long)blockIdx.x * 16;
  const int co0 = blockIdx.y * (NB * 16) + (lane & 15);  // channel of n-tile 0
  const int kb = (lane >> 4) * 8;                 // A-fragment K base per lane half

  // pixel this lane loads for the A fragment (M = lane & 15)
  const long long m = mbase + (lane & 15);
  const int hw = d.Hout * d.Wout;
  const int b  = (int)(m / hw);
  const int rm = (int)(m - (long long)b * hw);
  const int oy = rm / d.Wout;
  const int ox = rm - oy * d.Wout;

  const f16* __restrict__ wrow[NB];
#pragma unroll
  for (int nb = 0; nb < NB; ++nb)
    wrow[nb] = wpk + (long long)(co0 + nb * 16) * d.Kpad + (lane >> 4) * 8;

  const bool vec = (d.Cin & 31) == 0;

  v8f acc[NB];
#pragma unroll
  for (int nb = 0; nb < NB; ++nb) acc[nb] = (v8f){};

  for (int kc = 0; kc < d.Kpad; kc += 32) {
    // ---- A fragment (shared by all NB n-tiles) ----
    v16h a = {};
    if (vec) {
      const int tap = kc / d.Cin;                 // 32 | Cin -> chunk in one tap
      const int ci0 = kc - tap * d.Cin;
      const int ky = tap / d.KW;
      const int kx = tap - ky * d.KW;
      const int iy = oy * d.stride + ky - d.pad;
      const int ix = ox * d.stride + kx - d.pad;
      if ((unsigned)iy < (unsigned)d.Hin && (unsigned)ix < (unsigned)d.Win) {
        const f16* p = act + ((long long)((b * d.Hin + iy) * d.Win + ix)) * d.Cin + ci0 + kb;
        v8h lo = *(const v8h*)p;                  // K = kb .. kb+7
        v8h hi = *(const v8h*)(p + 16);           // K = kb+16 .. kb+23
        a = __builtin_shufflevector(lo, hi, 0,1,2,3,4,5,6,7,8,9,10,11,12,13,14,15);
      }
    } else {
      // generic scalar-gather path (conv1: Cin=3, Kreal=27 -> one padded chunk)
#pragma unroll
      for (int e = 0; e < 16; ++e) {
        const int j = e >> 1, hl = e & 1;
        const int krel = (j < 4) ? (kb + 2 * j + hl) : (kb + 16 + 2 * (j - 4) + hl);
        const int k = kc + krel;
        f16 v = (f16)0.f;
        if (k < d.Kreal) {
          const int ci  = k % d.Cin;
          const int tap = k / d.Cin;
          const int ky = tap / d.KW;
          const int kx = tap - ky * d.KW;
          const int iy = oy * d.stride + ky - d.pad;
          const int ix = ox * d.stride + kx - d.pad;
          if ((unsigned)iy < (unsigned)d.Hin && (unsigned)ix < (unsigned)d.Win)
            v = act[((long long)((b * d.Hin + iy) * d.Win + ix)) * d.Cin + ci];
        }
        a[e] = v;
      }
    }

    // prefetch next K-chunk of the packed weight rows (global_prefetch_b8)
    if (kc + 32 < d.Kpad) __builtin_prefetch(wrow[0] + kc + 32, 0, 0);

    // ---- NB independent WMMAs against the shared A fragment ----
#pragma unroll
    for (int nb = 0; nb < NB; ++nb) {
      // B fragment: lane column N = lane&15, K = kc + (lane>>4)*8 + e (e<8), +16 for e>=8
      const f16* bp = wrow[nb] + kc;
      v8h blo = *(const v8h*)bp;
      v8h bhi = *(const v8h*)(bp + 16);
      v16h bm = __builtin_shufflevector(blo, bhi, 0,1,2,3,4,5,6,7,8,9,10,11,12,13,14,15);
      acc[nb] = __builtin_amdgcn_wmma_f32_16x16x32_f16(false, a, false, bm,
                                                       (short)0, acc[nb], false, false);
    }
  }

  // fused epilogue: y = alpha*(acc + cbias) + beta, optional ReLU
#pragma unroll
  for (int nb = 0; nb < NB; ++nb) {
    const int co = co0 + nb * 16;
    const float alpha = d.hasG ? gamma[co] * BNSCALE : 1.0f;
    const float cb = d.hasCB ? cbias[co] : 0.0f;
    const float bt = d.hasBT ? beta[co] : 0.0f;
#pragma unroll
    for (int r = 0; r < 8; ++r) {
      const long long mm = mbase + (lane >> 4) * 8 + r;   // C/D layout: M=(L>>4)*8+r
      float v = alpha * (acc[nb][r] + cb) + bt;
      if (d.relu) v = fmaxf(v, 0.0f);
      const long long o = mm * d.Cout + co;
      if (out16) out16[o] = (f16)v;
      if (out32) out32[o] = v;
    }
  }
}

// NCHW f32 input -> NHWC f16
__global__ void xcvt_kernel(const float* __restrict__ x, f16* __restrict__ xh, int npix)
{
  int i = blockIdx.x * blockDim.x + threadIdx.x;
  if (i >= npix) return;
  const int HW = 768 * 768;
  int b = i / HW, p = i - b * HW;
#pragma unroll
  for (int c = 0; c < 3; ++c)
    xh[(long long)i * 3 + c] = (f16)x[(long long)(b * 3 + c) * HW + p];
}

// OIHW f32 -> [Cout][Kpad] f16, K=(ky*KW+kx)*Cin+ci, zero-padded
__global__ void wpack_kernel(const float* __restrict__ w, f16* __restrict__ wp,
                             int Cin, int KH, int KW, int Kpad, int total)
{
  int i = blockIdx.x * blockDim.x + threadIdx.x;
  if (i >= total) return;
  int co = i / Kpad, k = i - co * Kpad;
  float v = 0.0f;
  const int Kr = Cin * KH * KW;
  if (k < Kr) {
    int ci = k % Cin;
    int tap = k / Cin;
    int ky = tap / KW, kx = tap - ky * KW;
    v = w[((long long)(co * Cin + ci) * KH + ky) * KW + kx];
  }
  wp[i] = (f16)v;
}

// detector 1x1 conv 32 -> 1 (+bias), f32 score map
__global__ void det2_kernel(const f16* __restrict__ det1, const float* __restrict__ w2,
                            const float* __restrict__ b2, float* __restrict__ score, int M)
{
  int m = blockIdx.x * blockDim.x + threadIdx.x;
  if (m >= M) return;
  float s = b2[0];
#pragma unroll
  for (int c = 0; c < 32; ++c) s += (float)det1[(long long)m * 32 + c] * w2[c];
  score[m] = s;
}

// maxpool(4x4, stride 1, pad (2,1)x(2,1)) equality NMS
__global__ void nms_kernel(const float* __restrict__ score, float* __restrict__ nmsb, int n)
{
  int i = blockIdx.x * blockDim.x + threadIdx.x;
  if (i >= n) return;
  const int HW = 96 * 96;
  int b = i / HW, p = i - b * HW;
  int y = p / 96, x = p - y * 96;
  float s = score[i];
  float mx = -__builtin_huge_valf();
  for (int dy = -2; dy <= 1; ++dy) {
    int yy = y + dy;
    if ((unsigned)yy >= 96u) continue;
    for (int dx = -2; dx <= 1; ++dx) {
      int xx = x + dx;
      if ((unsigned)xx >= 96u) continue;
      float t = score[b * HW + yy * 96 + xx];
      mx = fmaxf(mx, t);
    }
  }
  nmsb[i] = (s == mx) ? s : 0.0f;
}

// exact sorted top-4096 per batch: bitonic sort of 16384 packed keys in LDS
// key = sortable(value) << 32 | ~idx  (descending value, ascending idx on ties)
__global__ __launch_bounds__(1024) void topk_kernel(
    const float* __restrict__ nmsb, float* __restrict__ out_kp,
    float* __restrict__ out_sc, unsigned* __restrict__ idxb, int HW, int Wd)
{
  extern __shared__ unsigned long long keys[];
  const int b = blockIdx.x;
  const int tid = threadIdx.x, nth = blockDim.x;
  const float* src = nmsb + (long long)b * HW;

  for (int i = tid; i < TOPK_N; i += nth) {
    float v = (i < HW) ? src[i] : -__builtin_huge_valf();
    unsigned u = __float_as_uint(v);
    u = (u & 0x80000000u) ? ~u : (u | 0x80000000u);
    keys[i] = ((unsigned long long)u << 32) | (unsigned)(~(unsigned)i);
  }
  __syncthreads();

  for (int size = 2; size <= TOPK_N; size <<= 1) {
    for (int stride = size >> 1; stride > 0; stride >>= 1) {
      for (int i = tid; i < TOPK_N / 2; i += nth) {
        int lo = (i << 1) - (i & (stride - 1));
        int hi = lo + stride;
        bool descend = ((lo & size) == 0);
        unsigned long long a = keys[lo], c = keys[hi];
        bool sw = descend ? (a < c) : (a > c);
        if (sw) { keys[lo] = c; keys[hi] = a; }
      }
      __syncthreads();
    }
  }

  for (int j = tid; j < NKPT; j += nth) {
    unsigned long long kk = keys[j];
    unsigned u = (unsigned)(kk >> 32);
    float v = (u & 0x80000000u) ? __uint_as_float(u ^ 0x80000000u) : __uint_as_float(~u);
    unsigned idx = ~((unsigned)kk);
    long long o = (long long)b * NKPT + j;
    out_sc[o] = v;
    out_kp[o * 2 + 0] = (float)(idx % (unsigned)Wd);
    out_kp[o * 2 + 1] = (float)(idx / (unsigned)Wd);
    idxb[o] = idx;
  }
}

// descriptor gather at top-k indices + L2 normalize
__global__ void gather_norm_kernel(const float* __restrict__ dmap,
                                   const unsigned* __restrict__ idxb,
                                   float* __restrict__ outd, int HW)
{
  int t = blockIdx.x * blockDim.x + threadIdx.x;
  if (t >= 4 * NKPT) return;
  int b = t / NKPT;
  unsigned p = idxb[t];
  const float* src = dmap + ((long long)b * HW + p) * 64;
  float s = 0.0f;
#pragma unroll
  for (int c = 0; c < 64; ++c) { float v = src[c]; s += v * v; }
  float nrm = fmaxf(sqrtf(s), 1e-12f);
  float inv = 1.0f / nrm;
  float* dst = outd + (long long)t * 64;
#pragma unroll
  for (int c = 0; c < 64; ++c) dst[c] = src[c] * inv;
}

// ---------------------------------------------------------------------------
extern "C" void kernel_launch(void* const* d_in, const int* in_sizes, int n_in,
                              void* d_out, int out_size, void* d_ws, size_t ws_size,
                              hipStream_t stream)
{
  (void)in_sizes; (void)n_in; (void)out_size; (void)ws_size;

  const float* x      = (const float*)d_in[0];
  const float* w1     = (const float*)d_in[1];
  const float* g1     = (const float*)d_in[2];
  const float* b1     = (const float*)d_in[3];
  const float* w2     = (const float*)d_in[4];
  const float* g2     = (const float*)d_in[5];
  const float* b2     = (const float*)d_in[6];
  const float* w3     = (const float*)d_in[7];
  const float* g3     = (const float*)d_in[8];
  const float* b3     = (const float*)d_in[9];
  const float* w4     = (const float*)d_in[10];
  const float* g4     = (const float*)d_in[11];
  const float* b4     = (const float*)d_in[12];
  const float* wf     = (const float*)d_in[13];
  const float* bfp    = (const float*)d_in[14];
  const float* det_w1 = (const float*)d_in[15];
  const float* det_b1 = (const float*)d_in[16];
  const float* det_g  = (const float*)d_in[17];
  const float* det_be = (const float*)d_in[18];
  const float* det_w2 = (const float*)d_in[19];
  const float* det_b2 = (const float*)d_in[20];
  const float* ds_w1  = (const float*)d_in[21];
  const float* ds_b1  = (const float*)d_in[22];
  const float* ds_g1  = (const float*)d_in[23];
  const float* ds_be1 = (const float*)d_in[24];
  const float* ds_w2  = (const float*)d_in[25];
  const float* ds_b2  = (const float*)d_in[26];
  const float* ds_g2  = (const float*)d_in[27];
  const float* ds_be2 = (const float*)d_in[28];

  const int M1 = 4 * 384 * 384;   // 589824
  const int M2 = 4 * 192 * 192;   // 147456
  const int M3 = 4 * 96 * 96;     // 36864
  const int HW = 96 * 96;         // 9216

  // ---- workspace carve (total ~120 MB, fits MI455X 192 MB L2) ----
  char* wsp = (char*)d_ws;
  auto carve = [&](size_t bytes) -> void* {
    void* p = (void*)wsp;
    wsp += (bytes + 255) & ~(size_t)255;
    return p;
  };
  f16*   xh   = (f16*)  carve((size_t)4 * 768 * 768 * 3 * 2);
  f16*   w1p  = (f16*)  carve((size_t)32 * 32 * 2);
  f16*   w2p  = (f16*)  carve((size_t)64 * 288 * 2);
  f16*   w3p  = (f16*)  carve((size_t)128 * 576 * 2);
  f16*   w4p  = (f16*)  carve((size_t)256 * 1152 * 2);
  f16*   wfp  = (f16*)  carve((size_t)64 * 256 * 2);
  f16*   dw1p = (f16*)  carve((size_t)32 * 576 * 2);
  f16*   sw1p = (f16*)  carve((size_t)32 * 576 * 2);
  f16*   sw2p = (f16*)  carve((size_t)64 * 32 * 2);
  f16*   c1   = (f16*)  carve((size_t)M1 * 32 * 2);
  f16*   c2   = (f16*)  carve((size_t)M2 * 64 * 2);
  f16*   c3   = (f16*)  carve((size_t)M3 * 128 * 2);
  f16*   c4   = (f16*)  carve((size_t)M3 * 256 * 2);
  f16*   feat = (f16*)  carve((size_t)M3 * 64 * 2);
  f16*   det1 = (f16*)  carve((size_t)M3 * 32 * 2);
  float* scr  = (float*)carve((size_t)M3 * 4);
  float* nmsb = (float*)carve((size_t)M3 * 4);
  f16*   ds1  = (f16*)  carve((size_t)M3 * 32 * 2);
  float* dmap = (float*)carve((size_t)M3 * 64 * 4);
  unsigned* idxb = (unsigned*)carve((size_t)4 * NKPT * 4);

  float* out_kp = (float*)d_out;                 // [4,4096,2]
  float* out_sc = out_kp + 4 * NKPT * 2;         // [4,4096]
  float* out_d  = out_sc + 4 * NKPT;             // [4,4096,64]

  // ---- input conversion + weight packing ----
  {
    int npix = 4 * 768 * 768;
    xcvt_kernel<<<(npix + 255) / 256, 256, 0, stream>>>(x, xh, npix);
  }
  auto pack = [&](const float* w, f16* wp, int Cout, int Cin, int KH, int KW, int Kpad) {
    int total = Cout * Kpad;
    wpack_kernel<<<(total + 255) / 256, 256, 0, stream>>>(w, wp, Cin, KH, KW, Kpad, total);
  };
  pack(w1,     w1p,  32,  3,   3, 3, 32);
  pack(w2,     w2p,  64,  32,  3, 3, 288);
  pack(w3,     w3p,  128, 64,  3, 3, 576);
  pack(w4,     w4p,  256, 128, 3, 3, 1152);
  pack(wf,     wfp,  64,  256, 1, 1, 256);
  pack(det_w1, dw1p, 32,  64,  3, 3, 576);
  pack(ds_w1,  sw1p, 32,  64,  3, 3, 576);
  pack(ds_w2,  sw2p, 64,  32,  1, 1, 32);

  // ---- convs (WMMA, register-blocked over output channels) ----
  auto conv = [&](const f16* act, const f16* wpk,
                  const float* g, const float* be, const float* cb,
                  f16* o16, float* o32, ConvDesc d) {
    int mtot = 4 * d.Hout * d.Wout;
    if (d.Cout >= 256) {
      dim3 grid(mtot / 16, d.Cout / 64);
      conv_wmma<4><<<grid, 32, 0, stream>>>(act, wpk, g, be, cb, o16, o32, d);
    } else {
      dim3 grid(mtot / 16, d.Cout / 32);
      conv_wmma<2><<<grid, 32, 0, stream>>>(act, wpk, g, be, cb, o16, o32, d);
    }
  };
  // c1 = CBR(x): 3->32, s2
  conv(xh, w1p, g1, b1, nullptr, c1, nullptr,
       {768, 768, 3, 384, 384, 32, 3, 3, 2, 1, 27, 32, 1, 0, 1, 1});
  // c2 = CBR(c1): 32->64, s2
  conv(c1, w2p, g2, b2, nullptr, c2, nullptr,
       {384, 384, 32, 192, 192, 64, 3, 3, 2, 1, 288, 288, 1, 0, 1, 1});
  // c3 = CBR(c2): 64->128, s2
  conv(c2, w3p, g3, b3, nullptr, c3, nullptr,
       {192, 192, 64, 96, 96, 128, 3, 3, 2, 1, 576, 576, 1, 0, 1, 1});
  // c4 = CBR(c3): 128->256, s1
  conv(c3, w4p, g4, b4, nullptr, c4, nullptr,
       {96, 96, 128, 96, 96, 256, 3, 3, 1, 1, 1152, 1152, 1, 0, 1, 1});
  // feat = conv1x1(c4) + bf : 256->64
  conv(c4, wfp, nullptr, nullptr, bfp, feat, nullptr,
       {96, 96, 256, 96, 96, 64, 1, 1, 1, 0, 256, 256, 0, 1, 0, 0});

  // ---- detector head ----
  conv(feat, dw1p, det_g, det_be, det_b1, det1, nullptr,
       {96, 96, 64, 96, 96, 32, 3, 3, 1, 1, 576, 576, 1, 1, 1, 1});
  det2_kernel<<<(M3 + 255) / 256, 256, 0, stream>>>(det1, det_w2, det_b2, scr, M3);
  nms_kernel<<<(M3 + 255) / 256, 256, 0, stream>>>(scr, nmsb, M3);
  topk_kernel<<<4, 1024, TOPK_N * sizeof(unsigned long long), stream>>>(
      nmsb, out_kp, out_sc, idxb, HW, 96);

  // ---- descriptor head ----
  conv(feat, sw1p, ds_g1, ds_be1, ds_b1, ds1, nullptr,
       {96, 96, 64, 96, 96, 32, 3, 3, 1, 1, 576, 576, 1, 1, 1, 1});
  conv(ds1, sw2p, ds_g2, ds_be2, ds_b2, nullptr, dmap,
       {96, 96, 32, 96, 96, 64, 1, 1, 1, 0, 32, 32, 1, 1, 1, 0});
  gather_norm_kernel<<<(4 * NKPT + 255) / 256, 256, 0, stream>>>(dmap, idxb, out_d, HW);
}
